// VisionMamba_83451214561747
// MI455X (gfx1250) — compile-verified, matrix operand order checked by reading, not run on hardware
//
#include <hip/hip_runtime.h>
#include <hip/hip_bf16.h>
#include <math.h>

typedef __attribute__((ext_vector_type(16))) _Float16 v16h;
typedef __attribute__((ext_vector_type(8)))  float    v8f;

#define LSEQ    197
#define NB      4
#define NT      (NB * LSEQ)     // 788 tokens
#define DM      192             // D_MODEL
#define DI      384             // D_INNER
#define DSTATE  16
#define DTR     12              // DT_RANK
#define NXP     44              // DT_RANK + 2*D_STATE
#define NPATCHK 196
#define TOKPOSK 98
#define NCLSK   1000

// ---------------------------------------------------------------- activations
__device__ __forceinline__ float siluf(float x) {
  return x / (1.0f + __expf(-x));
}
__device__ __forceinline__ float softplusf(float x) {
  return (x > 20.0f) ? x : log1pf(__expf(x));
}

// ---------------------------------------------------------------- WMMA GEMM
// C[m,n] = sum_k A[m,k] * Bw[n,k] + bias[n]
// A: M x K (lda), Bw: N x K (ldb), both f32 (converted to f16 on load).
// Requirements: K % 64 == 0; lda, ldb multiples of 4 (16B-aligned rows).
// One wave per 16x16 output tile. Row/col CLAMPED for loads (always-valid
// memory, no divergence); guards only on stores. Two WMMAs per K iteration.
__device__ __forceinline__ v16h frag_from(const float* __restrict__ row, int kb)
{
  const float4 x0 = *(const float4*)(row + kb);
  const float4 x1 = *(const float4*)(row + kb + 4);
  const float4 x2 = *(const float4*)(row + kb + 16);
  const float4 x3 = *(const float4*)(row + kb + 20);
  v16h f;
  f[0]  = (_Float16)x0.x; f[1]  = (_Float16)x0.y;
  f[2]  = (_Float16)x0.z; f[3]  = (_Float16)x0.w;
  f[4]  = (_Float16)x1.x; f[5]  = (_Float16)x1.y;
  f[6]  = (_Float16)x1.z; f[7]  = (_Float16)x1.w;
  f[8]  = (_Float16)x2.x; f[9]  = (_Float16)x2.y;
  f[10] = (_Float16)x2.z; f[11] = (_Float16)x2.w;
  f[12] = (_Float16)x3.x; f[13] = (_Float16)x3.y;
  f[14] = (_Float16)x3.z; f[15] = (_Float16)x3.w;
  return f;
}

__global__ __launch_bounds__(32)
void gemm_wmma_f16(const float* __restrict__ A, int lda,
                   const float* __restrict__ Bw, int ldb,
                   const float* __restrict__ bias,
                   float* __restrict__ C, int ldc,
                   int M, int N, int K)
{
  const int lane   = threadIdx.x;
  const int tilesN = (N + 15) >> 4;
  const int tn     = (blockIdx.x % tilesN) << 4;
  const int tm     = (blockIdx.x / tilesN) << 4;

  const int mrow  = tm + (lane & 15);
  const int ncol  = tn + (lane & 15);
  const int khalf = (lane >> 4) << 3;   // 0 for lanes 0-15, 8 for lanes 16-31

  // clamp to valid rows: out-of-range M/N rows only feed discarded outputs
  const int mc = (mrow < M) ? mrow : (M - 1);
  const int nc = (ncol < N) ? ncol : (N - 1);
  const float* __restrict__ arow = A  + (size_t)mc * lda + khalf;
  const float* __restrict__ brow = Bw + (size_t)nc * ldb + khalf;

  v8f acc = {};
  for (int kb = 0; kb < K; kb += 64) {
    if (kb + 64 < K) {              // prefetch next chunk -> global_prefetch_b8
      __builtin_prefetch(arow + kb + 64, 0, 1);
      __builtin_prefetch(brow + kb + 64, 0, 1);
    }
    const v16h a0 = frag_from(arow, kb);
    const v16h b0 = frag_from(brow, kb);
    acc = __builtin_amdgcn_wmma_f32_16x16x32_f16(false, a0, false, b0,
                                                 (short)0, acc, false, false);
    const v16h a1 = frag_from(arow, kb + 32);
    const v16h b1 = frag_from(brow, kb + 32);
    acc = __builtin_amdgcn_wmma_f32_16x16x32_f16(false, a1, false, b1,
                                                 (short)0, acc, false, false);
  }

  if (ncol < N) {
    const float bv = bias ? bias[ncol] : 0.0f;
    const int rbase = tm + ((lane >> 4) << 3);  // M=r (lanes 0-15), M=r+8 (16-31)
    #pragma unroll
    for (int r = 0; r < 8; ++r) {
      const int row = rbase + r;
      if (row < M) C[(size_t)row * ldc + ncol] = acc[r] + bv;
    }
  }
}

// ---------------------------------------------------------------- dt proj
// dt[t,d] = softplus( sum_{r<12} xdbl[t,r] * w[d,r] + bias[d] )  (K=12: VALU)
__global__ void dtproj_kernel(const float* __restrict__ xdbl,  // (NT, NXP)
                              const float* __restrict__ w,     // (DI, DTR)
                              const float* __restrict__ bias,  // (DI)
                              float* __restrict__ dt)          // (NT, DI)
{
  const int idx = blockIdx.x * blockDim.x + threadIdx.x;
  if (idx >= NT * DI) return;
  const int d = idx % DI, t = idx / DI;
  const float* xr = xdbl + (size_t)t * NXP;
  const float* wr = w    + (size_t)d * DTR;
  float acc = bias[d];
  #pragma unroll
  for (int r = 0; r < DTR; ++r) acc += xr[r] * wr[r];
  dt[idx] = softplusf(acc);
}

// ---------------------------------------------------------------- patch embed
__global__ void im2col_kernel(const float* __restrict__ x, float* __restrict__ col)
{
  const int idx = blockIdx.x * blockDim.x + threadIdx.x;
  if (idx >= 784 * 768) return;
  const int k = idx % 768, m = idx / 768;
  const int b = m / NPATCHK, ptc = m % NPATCHK;
  const int c = k >> 8, rem = k & 255, ph = rem >> 4, pw = rem & 15;
  const int py = ptc / 14, px = ptc % 14;
  col[idx] = x[(((size_t)b * 3 + c) * 224 + (py * 16 + ph)) * 224 + (px * 16 + pw)];
}

// tokens: [patch 0..97], cls, [patch 98..195]; + pos_embed. res initialized 0.
__global__ void assemble_kernel(const float* __restrict__ pout,
                                const float* __restrict__ cls,
                                const float* __restrict__ pos,
                                float* __restrict__ hid, float* __restrict__ res)
{
  const int idx = blockIdx.x * blockDim.x + threadIdx.x;
  if (idx >= NT * DM) return;
  const int d = idx % DM;
  const int t = (idx / DM) % LSEQ;
  const int b = idx / (DM * LSEQ);
  float v;
  if (t == TOKPOSK) v = cls[d];
  else {
    const int pidx = (t < TOKPOSK) ? t : (t - 1);
    v = pout[(size_t)(b * NPATCHK + pidx) * DM + d];
  }
  hid[idx] = v + pos[t * DM + d];
  res[idx] = 0.0f;
}

// ---------------------------------------------------------------- rmsnorm
// res += hid (in place); hn = rms(res) * w.  One block per token.
__global__ __launch_bounds__(64)
void rms_kernel(float* __restrict__ res, const float* __restrict__ hid,
                const float* __restrict__ w, float* __restrict__ hn)
{
  const int t = blockIdx.x, tid = threadIdx.x;
  __shared__ float ssum[64];
  float v[3]; float s = 0.0f;
  #pragma unroll
  for (int i = 0; i < 3; ++i) {
    const int d = tid * 3 + i;
    const float r = res[t * DM + d] + hid[t * DM + d];
    v[i] = r; res[t * DM + d] = r; s += r * r;
  }
  ssum[tid] = s; __syncthreads();
  for (int off = 32; off > 0; off >>= 1) {
    if (tid < off) ssum[tid] += ssum[tid + off];
    __syncthreads();
  }
  const float inv = rsqrtf(ssum[0] / (float)DM + 1e-5f);
  #pragma unroll
  for (int i = 0; i < 3; ++i) {
    const int d = tid * 3 + i;
    hn[t * DM + d] = v[i] * inv * w[d];
  }
}

// ---------------------------------------------------------------- causal conv
// Forward dir in natural order; backward dir computed directly in reversed
// coordinates (xc_b[t] corresponds to flipped-sequence position t).
__global__ void conv_silu_kernel(const float* __restrict__ xz,
                                 const float* __restrict__ wf, const float* __restrict__ bfv,
                                 const float* __restrict__ wb, const float* __restrict__ bbv,
                                 float* __restrict__ xcf, float* __restrict__ xcb)
{
  const int idx = blockIdx.x * blockDim.x + threadIdx.x;
  if (idx >= NT * DI) return;
  const int d = idx % DI;
  const int t = (idx / DI) % LSEQ;
  const int b = idx / (DI * LSEQ);
  float accf = bfv[d], accb = bbv[d];
  #pragma unroll
  for (int j = 0; j < 4; ++j) {
    const int s = t - 3 + j;
    if (s >= 0) {
      accf += wf[d * 4 + j] * xz[(size_t)(b * LSEQ + s) * 768 + d];
      accb += wb[d * 4 + j] * xz[(size_t)(b * LSEQ + (LSEQ - 1 - s)) * 768 + d];
    }
  }
  xcf[idx] = siluf(accf);
  xcb[idx] = siluf(accb);
}

// ---------------------------------------------------------------- SSM scan
// One block per batch; thread d owns channel d, 16-wide state in registers.
// B/C (shared across channels) broadcast through LDS each step.
__global__ __launch_bounds__(DI)
void ssm_scan_kernel(const float* __restrict__ xc,    // (NT, DI)
                     const float* __restrict__ dt,    // (NT, DI)
                     const float* __restrict__ xdbl,  // (NT, NXP)
                     const float* __restrict__ xz,    // (NT, 768): z at [384+d]
                     const float* __restrict__ A_log, // (DI, DSTATE)
                     const float* __restrict__ Dp,    // (DI)
                     float* __restrict__ y,           // (NT, DI)
                     int reversed_z)
{
  const int b = blockIdx.x;
  const int d = threadIdx.x;
  __shared__ float sB[DSTATE], sC[DSTATE];

  float Av[DSTATE];
  #pragma unroll
  for (int n = 0; n < DSTATE; ++n) Av[n] = -__expf(A_log[d * DSTATE + n]);
  const float Dv = Dp[d];

  float h[DSTATE];
  #pragma unroll
  for (int n = 0; n < DSTATE; ++n) h[n] = 0.0f;

  for (int t = 0; t < LSEQ; ++t) {
    const int tt = b * LSEQ + t;
    __syncthreads();
    if (d < DSTATE)          sB[d]          = xdbl[tt * NXP + DTR + d];
    else if (d < 2 * DSTATE) sC[d - DSTATE] = xdbl[tt * NXP + DTR + d];
    __syncthreads();

    const float dtv = dt[tt * DI + d];
    const float xcv = xc[tt * DI + d];
    const float dbx = dtv * xcv;
    float accv = 0.0f;
    #pragma unroll
    for (int n = 0; n < DSTATE; ++n) {
      const float dA = __expf(dtv * Av[n]);
      h[n] = dA * h[n] + dbx * sB[n];
      accv += h[n] * sC[n];
    }
    const int tz = reversed_z ? (b * LSEQ + (LSEQ - 1 - t)) : tt;
    const float zv = xz[(size_t)tz * 768 + DI + d];
    y[(size_t)tt * DI + d] = (accv + xcv * Dv) * siluf(zv);
  }
}

// yo[t] = 0.5 * (yf[t] + yb[L-1-t])   (un-flip backward direction)
__global__ void combine_kernel(const float* __restrict__ yf,
                               const float* __restrict__ yb,
                               float* __restrict__ yo)
{
  const int idx = blockIdx.x * blockDim.x + threadIdx.x;
  if (idx >= NT * DI) return;
  const int d = idx % DI;
  const int t = (idx / DI) % LSEQ;
  const int b = idx / (DI * LSEQ);
  yo[idx] = 0.5f * (yf[idx] + yb[(size_t)(b * LSEQ + (LSEQ - 1 - t)) * DI + d]);
}

// final feature: rms(res + hid) * w at token TOKPOS, per batch
__global__ __launch_bounds__(64)
void feat_kernel(const float* __restrict__ res, const float* __restrict__ hid,
                 const float* __restrict__ w, float* __restrict__ feat)
{
  const int b = blockIdx.x, tid = threadIdx.x;
  const int t = b * LSEQ + TOKPOSK;
  __shared__ float ssum[64];
  float v[3]; float s = 0.0f;
  #pragma unroll
  for (int i = 0; i < 3; ++i) {
    const int d = tid * 3 + i;
    const float r = res[t * DM + d] + hid[t * DM + d];
    v[i] = r; s += r * r;
  }
  ssum[tid] = s; __syncthreads();
  for (int off = 32; off > 0; off >>= 1) {
    if (tid < off) ssum[tid] += ssum[tid + off];
    __syncthreads();
  }
  const float inv = rsqrtf(ssum[0] / (float)DM + 1e-5f);
  #pragma unroll
  for (int i = 0; i < 3; ++i) {
    const int d = tid * 3 + i;
    feat[b * DM + d] = v[i] * inv * w[d];
  }
}

// ================================================================ host side
static inline int tiles16(int v) { return (v + 15) >> 4; }

extern "C" void kernel_launch(void* const* d_in, const int* in_sizes, int n_in,
                              void* d_out, int out_size, void* d_ws, size_t ws_size,
                              hipStream_t stream)
{
  const float* x         = (const float*)d_in[0];
  const float* patch_w   = (const float*)d_in[1];
  const float* patch_b   = (const float*)d_in[2];
  const float* cls_token = (const float*)d_in[3];
  const float* pos_embed = (const float*)d_in[4];
  const float* ln_w      = (const float*)d_in[5];
  const float* in_proj_w = (const float*)d_in[6];
  const float* convf_w   = (const float*)d_in[7];
  const float* convf_b   = (const float*)d_in[8];
  const float* convb_w   = (const float*)d_in[9];
  const float* convb_b   = (const float*)d_in[10];
  const float* xprojf_w  = (const float*)d_in[11];
  const float* xprojb_w  = (const float*)d_in[12];
  const float* dtprojf_w = (const float*)d_in[13];
  const float* dtprojf_b = (const float*)d_in[14];
  const float* dtprojb_w = (const float*)d_in[15];
  const float* dtprojb_b = (const float*)d_in[16];
  const float* A_logf    = (const float*)d_in[17];
  const float* A_logb    = (const float*)d_in[18];
  const float* Df        = (const float*)d_in[19];
  const float* Db        = (const float*)d_in[20];
  const float* out_proj_w= (const float*)d_in[21];
  const float* normf_w   = (const float*)d_in[22];
  const float* head_w    = (const float*)d_in[23];
  const float* head_b    = (const float*)d_in[24];
  float* out = (float*)d_out;

  // ---- carve workspace (~16 MB of f32)
  float* p = (float*)d_ws;
  auto carve = [&](size_t n) { float* q = p; p += n; return q; };
  float* im2col = carve((size_t)784 * 768);
  float* pout   = carve((size_t)784 * DM);
  float* hid    = carve((size_t)NT * DM);
  float* res    = carve((size_t)NT * DM);
  float* hn     = carve((size_t)NT * DM);
  float* xzbuf  = carve((size_t)NT * 768);
  float* xcf    = carve((size_t)NT * DI);
  float* xcb    = carve((size_t)NT * DI);
  float* xdf    = carve((size_t)NT * NXP);
  float* xdb    = carve((size_t)NT * NXP);
  float* dtf    = carve((size_t)NT * DI);
  float* dtb    = carve((size_t)NT * DI);
  float* yf     = carve((size_t)NT * DI);
  float* yb     = carve((size_t)NT * DI);
  float* ycomb  = carve((size_t)NT * DI);
  float* feat   = carve((size_t)NB * DM);

  // ---- patch embed: im2col + WMMA GEMM (784x768 @ 768x192)
  im2col_kernel<<<(784 * 768 + 255) / 256, 256, 0, stream>>>(x, im2col);
  gemm_wmma_f16<<<tiles16(784) * tiles16(DM), 32, 0, stream>>>(
      im2col, 768, patch_w, 768, patch_b, pout, DM, 784, DM, 768);
  assemble_kernel<<<(NT * DM + 255) / 256, 256, 0, stream>>>(
      pout, cls_token, pos_embed, hid, res);

  const int tM = tiles16(NT); // 50

  for (int l = 0; l < 24; ++l) {
    const float* lw  = ln_w      + (size_t)l * DM;
    const float* iw  = in_proj_w + (size_t)l * 2 * DI * DM;
    const float* cfw = convf_w   + (size_t)l * DI * 4;
    const float* cfb = convf_b   + (size_t)l * DI;
    const float* cbw = convb_w   + (size_t)l * DI * 4;
    const float* cbb = convb_b   + (size_t)l * DI;
    const float* xfw = xprojf_w  + (size_t)l * NXP * DI;
    const float* xbw = xprojb_w  + (size_t)l * NXP * DI;
    const float* dfw = dtprojf_w + (size_t)l * DI * DTR;
    const float* dfb = dtprojf_b + (size_t)l * DI;
    const float* dbw = dtprojb_w + (size_t)l * DI * DTR;
    const float* dbb = dtprojb_b + (size_t)l * DI;
    const float* alf = A_logf    + (size_t)l * DI * DSTATE;
    const float* alb = A_logb    + (size_t)l * DI * DSTATE;
    const float* dpf = Df        + (size_t)l * DI;
    const float* dpb = Db        + (size_t)l * DI;
    const float* ow  = out_proj_w+ (size_t)l * DM * DI;

    // res += hid; hn = rms(res) * lw
    rms_kernel<<<NT, 64, 0, stream>>>(res, hid, lw, hn);
    // xz = hn @ iw^T  (788x192 @ 192x768, K=192)
    gemm_wmma_f16<<<tM * tiles16(2 * DI), 32, 0, stream>>>(
        hn, DM, iw, DM, nullptr, xzbuf, 2 * DI, NT, 2 * DI, DM);
    // depthwise causal conv + silu, both directions
    conv_silu_kernel<<<(NT * DI + 255) / 256, 256, 0, stream>>>(
        xzbuf, cfw, cfb, cbw, cbb, xcf, xcb);
    // x_proj: (788x384 @ 384x44, K=384), both directions
    gemm_wmma_f16<<<tM * tiles16(NXP), 32, 0, stream>>>(
        xcf, DI, xfw, DI, nullptr, xdf, NXP, NT, NXP, DI);
    gemm_wmma_f16<<<tM * tiles16(NXP), 32, 0, stream>>>(
        xcb, DI, xbw, DI, nullptr, xdb, NXP, NT, NXP, DI);
    // dt = softplus(xdbl[:, :12] @ dt_w^T + dt_b)  (K=12 -> VALU kernel)
    dtproj_kernel<<<(NT * DI + 255) / 256, 256, 0, stream>>>(xdf, dfw, dfb, dtf);
    dtproj_kernel<<<(NT * DI + 255) / 256, 256, 0, stream>>>(xdb, dbw, dbb, dtb);
    // selective scan (forward + reversed)
    ssm_scan_kernel<<<NB, DI, 0, stream>>>(xcf, dtf, xdf, xzbuf, alf, dpf, yf, 0);
    ssm_scan_kernel<<<NB, DI, 0, stream>>>(xcb, dtb, xdb, xzbuf, alb, dpb, yb, 1);
    // ycomb = 0.5*(yf + flip(yb))
    combine_kernel<<<(NT * DI + 255) / 256, 256, 0, stream>>>(yf, yb, ycomb);
    // hid = ycomb @ ow^T  (788x384 @ 384x192, K=384)
    gemm_wmma_f16<<<tM * tiles16(DM), 32, 0, stream>>>(
        ycomb, DI, ow, DI, nullptr, hid, DM, NT, DM, DI);
  }

  // final norm at cls-token position, then head (4x192 @ 192x1000, K=192)
  feat_kernel<<<NB, 64, 0, stream>>>(res, hid, normf_w, feat);
  gemm_wmma_f16<<<tiles16(NB) * tiles16(NCLSK), 32, 0, stream>>>(
      feat, DM, head_w, DM, head_b, out, NCLSK, NB, NCLSK, DM);
}
